// SpatialLocalMultiheadAttention_39719857553699
// MI455X (gfx1250) — matrix operand order repaired and checked by reading, not compiled, not run on hardware
//
#include <hip/hip_runtime.h>

typedef _Float16 h16 __attribute__((ext_vector_type(16)));
typedef _Float16 h8  __attribute__((ext_vector_type(8)));
typedef _Float16 h4  __attribute__((ext_vector_type(4)));
typedef float    f8  __attribute__((ext_vector_type(8)));

#define AST 520   // x-window LDS row stride in halves (conflict-free across 16 rows)
#define HST 72    // per-head Q/K/Vt/P row stride in halves
#define CST 136   // ctx-pair row stride in halves

__device__ __forceinline__ f8 wmma_f16(h16 a, h16 b, f8 c) {
  // v_wmma_f32_16x16x32_f16  D = A*B + C
  return __builtin_amdgcn_wmma_f32_16x16x32_f16(false, a, false, b, (short)0, c, false, false);
}

// A-fragment (16x32 f16): lane m&15 = row; 16 contiguous K halves at kbase + 16*(lane>>4)
__device__ __forceinline__ h16 ldA(const _Float16* p) {
  h8 lo = *(const h8*)(p);
  h8 hi = *(const h8*)(p + 8);
  return __builtin_shufflevector(lo, hi, 0,1,2,3,4,5,6,7,8,9,10,11,12,13,14,15);
}
// B-fragment (32x16 f16): lane n&15 = col; 16 contiguous K halves at kbase + 16*(lane>>4)
__device__ __forceinline__ h16 ldB(const _Float16* p) {
  h8 lo = *(const h8*)(p);
  h8 hi = *(const h8*)(p + 8);
  return __builtin_shufflevector(lo, hi, 0,1,2,3,4,5,6,7,8,9,10,11,12,13,14,15);
}

// ---------------- small prep kernels ----------------

// f32 -> f16 weight conversion (W_in: 1536x512, W_out: 512x512)
__global__ void cvt_kernel(const float* __restrict__ Wi, const float* __restrict__ Wo,
                           _Float16* __restrict__ Wih, _Float16* __restrict__ Woh) {
  int idx = blockIdx.x * 256 + threadIdx.x;
  if (idx < 786432) {
    Wih[idx] = (_Float16)Wi[idx];
  } else {
    int j = idx - 786432;
    if (j < 262144) Woh[j] = (_Float16)Wo[j];
  }
}

// posqk[s][j] = sum_c pos[s][c] * W_in[j][c] + b_in[j],  s<64, j<1024 (covers Wq and Wk rows)
__global__ void posqk_kernel(const float* __restrict__ pos, const float* __restrict__ W_in,
                             const float* __restrict__ b_in, float* __restrict__ posqk) {
  int idx = blockIdx.x * 256 + threadIdx.x;   // 65536 threads
  int s = idx >> 10, j = idx & 1023;
  const float4* p = (const float4*)(pos + s * 512);
  const float4* w = (const float4*)(W_in + (size_t)j * 512);
  float acc = 0.f;
  for (int i = 0; i < 128; ++i) {
    float4 a = p[i], bb = w[i];
    acc += a.x * bb.x + a.y * bb.y + a.z * bb.z + a.w * bb.w;
  }
  posqk[idx] = acc + b_in[j];
}

// ---------------- fused attention kernel: 1 block = 1 window (64 tokens) ----------------

__global__ __launch_bounds__(256, 1)
void fused_attn(const float* __restrict__ x, const float* __restrict__ posqk,
                const float* __restrict__ b_in, const float* __restrict__ b_out,
                const _Float16* __restrict__ Wih, const _Float16* __restrict__ Woh,
                float* __restrict__ out) {
  extern __shared__ _Float16 smem[];
  _Float16* Axw = smem;                 // 64 x AST : x window f16
  _Float16* Q2  = Axw + 64 * AST;       // 2 heads x 64 tok x HST (token-major)
  _Float16* K2  = Q2 + 2 * 64 * HST;    // 2 heads x 64 tok x HST (token-major)
  _Float16* Vt  = K2 + 2 * 64 * HST;    // 2 heads x 64 d   x HST (d-major, token inner)
  _Float16* Ps  = Vt + 2 * 64 * HST;    // per-wave 16 x HST softmax staging
  _Float16* Cx  = Ps + 8 * 16 * HST;    // 64 tok x CST ctx for current head-pair

  const int tid = threadIdx.x;
  const int lane = tid & 31, wv = tid >> 5;
  const int l15 = lane & 15, hi = lane >> 4;
  const int b = blockIdx.x;
  const int iww = b & 7, ihh = (b >> 3) & 7, ntt = b >> 6;
  const int pixbase = ntt * 4096 + ihh * 8 * 64 + iww * 8; // + (s>>3)*64 + (s&7)

  // ---- phase 1: load 64x512 f32 window -> f16 LDS ----
  {
    int row = tid >> 2, part = tid & 3;
    size_t goff = (size_t)(pixbase + (row >> 3) * 64 + (row & 7)) * 512;
    const float4* src = (const float4*)(x + goff);
    _Float16* dst = &Axw[row * AST];
    #pragma unroll
    for (int i = 0; i < 32; ++i) {
      int c4 = part + i * 4;
      float4 v = src[c4];
      h4 hv; hv[0] = (_Float16)v.x; hv[1] = (_Float16)v.y; hv[2] = (_Float16)v.z; hv[3] = (_Float16)v.w;
      *(h4*)(dst + c4 * 4) = hv;
    }
  }
  __syncthreads();

  const f8 zf = {};
  f8 oacc[4][4];   // persistent out-projection accumulators: [n-subtile][m-tile]
  #pragma unroll
  for (int i = 0; i < 4; ++i)
    #pragma unroll
    for (int m = 0; m < 4; ++m) oacc[i][m] = zf;

  const int hl  = wv >> 2;  // head within pair
  const int dt4 = wv & 3;   // d/n/m sub-tile index for this wave

  for (int hp = 0; hp < 4; ++hp) {        // head pairs
    const int head = hp * 2 + hl;

    // ---- phase 2: QKV projection for this head pair ----
    // ks-outer loop: each x-window A-fragment is loaded once and reused for q, k and v
    {
      const size_t rq = (size_t)(0   + head * 64 + dt4 * 16 + l15) * 512 + 16 * hi;
      const size_t rk = (size_t)(512 + head * 64 + dt4 * 16 + l15) * 512 + 16 * hi;
      const size_t rv = (size_t)(1024 + head * 64 + dt4 * 16 + l15) * 512 + 16 * hi;
      const _Float16* wpq = Wih + rq;
      const _Float16* wpk = Wih + rk;
      const _Float16* wpv = Wih + rv;

      f8 qa[4], ka[4], va[4];
      #pragma unroll
      for (int m = 0; m < 4; ++m) { qa[m] = zf; ka[m] = zf; va[m] = zf; }

      #pragma unroll
      for (int ks = 0; ks < 16; ++ks) {
        h16 af[4];
        #pragma unroll
        for (int mt = 0; mt < 4; ++mt)
          af[mt] = ldA(&Axw[(mt * 16 + l15) * AST + ks * 32 + 16 * hi]);
        h16 bq = ldB(wpq + ks * 32);
        #pragma unroll
        for (int mt = 0; mt < 4; ++mt) qa[mt] = wmma_f16(af[mt], bq, qa[mt]);
        h16 bk = ldB(wpk + ks * 32);
        #pragma unroll
        for (int mt = 0; mt < 4; ++mt) ka[mt] = wmma_f16(af[mt], bk, ka[mt]);
        h16 bv = ldB(wpv + ks * 32);
        #pragma unroll
        for (int mt = 0; mt < 4; ++mt) va[mt] = wmma_f16(af[mt], bv, va[mt]);
      }

      // epilogue: q -> Q2 (with posqk+bias, *1/sqrt(64)), k -> K2 (with posqk+bias), v -> Vt (d-major)
      {
        const int jq = head * 64 + dt4 * 16 + l15;
        _Float16* qd = Q2 + hl * (64 * HST);
        _Float16* kd = K2 + hl * (64 * HST);
        #pragma unroll
        for (int mt = 0; mt < 4; ++mt) {
          #pragma unroll
          for (int r = 0; r < 8; ++r) {
            int s = mt * 16 + 8 * hi + r;
            float vq = (qa[mt][r] + posqk[s * 1024 + jq]) * 0.125f;
            float vk =  ka[mt][r] + posqk[s * 1024 + 512 + jq];
            qd[s * HST + dt4 * 16 + l15] = (_Float16)vq;
            kd[s * HST + dt4 * 16 + l15] = (_Float16)vk;
          }
        }
        float bv = b_in[1024 + head * 64 + dt4 * 16 + l15];
        #pragma unroll
        for (int mt = 0; mt < 4; ++mt) {
          _Float16* vd = &Vt[hl * (64 * HST) + (dt4 * 16 + l15) * HST + mt * 16 + 8 * hi];
          #pragma unroll
          for (int r = 0; r < 8; ++r) vd[r] = (_Float16)(va[mt][r] + bv);
        }
      }
    }
    __syncthreads();

    // ---- phase 3: attention (wave -> head hl, query m-tile dt4) ----
    {
      const int mt = dt4;
      const _Float16* Qb = Q2 + hl * (64 * HST);
      const _Float16* Kb = K2 + hl * (64 * HST);
      const _Float16* Vb = Vt + hl * (64 * HST);
      f8 sacc[4];
      #pragma unroll
      for (int k = 0; k < 4; ++k) sacc[k] = zf;
      #pragma unroll
      for (int ks = 0; ks < 2; ++ks) {
        h16 af = ldA(Qb + (mt * 16 + l15) * HST + ks * 32 + 16 * hi);
        h16 bf[4];
        #pragma unroll
        for (int kt = 0; kt < 4; ++kt)
          bf[kt] = ldB(Kb + (kt * 16 + l15) * HST + ks * 32 + 16 * hi);
        #pragma unroll
        for (int kt = 0; kt < 4; ++kt) sacc[kt] = wmma_f16(af, bf[kt], sacc[kt]);
      }
      _Float16* Pw = Ps + wv * (16 * HST);
      #pragma unroll
      for (int r = 0; r < 8; ++r) {
        float m = -1e30f;
        #pragma unroll
        for (int kt = 0; kt < 4; ++kt) m = fmaxf(m, sacc[kt][r]);
        #pragma unroll
        for (int off = 8; off >= 1; off >>= 1) m = fmaxf(m, __shfl_xor(m, off, 32));
        float e[4], sum = 0.f;
        #pragma unroll
        for (int kt = 0; kt < 4; ++kt) { e[kt] = __expf(sacc[kt][r] - m); sum += e[kt]; }
        #pragma unroll
        for (int off = 8; off >= 1; off >>= 1) sum += __shfl_xor(sum, off, 32);
        float rinv = 1.0f / sum;
        #pragma unroll
        for (int kt = 0; kt < 4; ++kt)
          Pw[(8 * hi + r) * HST + kt * 16 + l15] = (_Float16)(e[kt] * rinv);
      }
      f8 cacc[4];
      #pragma unroll
      for (int k = 0; k < 4; ++k) cacc[k] = zf;
      #pragma unroll
      for (int ks = 0; ks < 2; ++ks) {
        h16 af = ldA(Pw + l15 * HST + ks * 32 + 16 * hi);
        h16 bf[4];
        #pragma unroll
        for (int dt = 0; dt < 4; ++dt)
          bf[dt] = ldB(Vb + (dt * 16 + l15) * HST + ks * 32 + 16 * hi);
        #pragma unroll
        for (int dt = 0; dt < 4; ++dt) cacc[dt] = wmma_f16(af, bf[dt], cacc[dt]);
      }
      #pragma unroll
      for (int dt = 0; dt < 4; ++dt)
        #pragma unroll
        for (int r = 0; r < 8; ++r)
          Cx[(mt * 16 + 8 * hi + r) * CST + hl * 64 + dt * 16 + l15] = (_Float16)cacc[dt][r];
    }
    __syncthreads();

    // ---- phase 4: partial out-projection (accumulate over head pairs) ----
    #pragma unroll
    for (int ks = 0; ks < 4; ++ks) {
      h16 af[4];
      #pragma unroll
      for (int mt = 0; mt < 4; ++mt)
        af[mt] = ldA(&Cx[(mt * 16 + l15) * CST + ks * 32 + 16 * hi]);
      #pragma unroll
      for (int i = 0; i < 4; ++i) {
        const _Float16* wp = Woh + (size_t)(wv * 64 + i * 16 + l15) * 512 + hp * 128 + ks * 32 + 16 * hi;
        h16 bf = ldB(wp);
        #pragma unroll
        for (int mt = 0; mt < 4; ++mt) oacc[i][mt] = wmma_f16(af[mt], bf, oacc[i][mt]);
      }
    }
    // no barrier needed here: next phase-2 writes Q2/K2/Vt, whose last readers are
    // barrier-separated (post-phase-3 barrier of this iteration).
  }

  // ---- final: bias + store f32 to (N,T,H,W,C) layout ----
  #pragma unroll
  for (int i = 0; i < 4; ++i) {
    float bo = b_out[wv * 64 + i * 16 + l15];
    #pragma unroll
    for (int mt = 0; mt < 4; ++mt) {
      #pragma unroll
      for (int r = 0; r < 8; ++r) {
        int s = mt * 16 + 8 * hi + r;
        size_t off = (size_t)(pixbase + (s >> 3) * 64 + (s & 7)) * 512 + wv * 64 + i * 16 + l15;
        out[off] = oacc[i][mt][r] + bo;
      }
    }
  }
}

// ---------------- host launch ----------------

extern "C" void kernel_launch(void* const* d_in, const int* in_sizes, int n_in,
                              void* d_out, int out_size, void* d_ws, size_t ws_size,
                              hipStream_t stream) {
  const float* x     = (const float*)d_in[0];
  const float* pos   = (const float*)d_in[1];
  const float* W_in  = (const float*)d_in[2];
  const float* b_in  = (const float*)d_in[3];
  const float* W_out = (const float*)d_in[4];
  const float* b_out = (const float*)d_in[5];
  float* out = (float*)d_out;

  _Float16* Wih = (_Float16*)d_ws;          // 1536*512 halves
  _Float16* Woh = Wih + 786432;             // 512*512 halves
  float* posqk  = (float*)(Woh + 262144);   // 64*1024 f32

  const int lds_bytes = (64 * AST + 6 * 64 * HST + 8 * 16 * HST + 64 * CST) * 2; // 157,696 B
  (void)hipFuncSetAttribute((const void*)fused_attn,
                            hipFuncAttributeMaxDynamicSharedMemorySize, lds_bytes);

  cvt_kernel<<<4096, 256, 0, stream>>>(W_in, W_out, Wih, Woh);
  posqk_kernel<<<256, 256, 0, stream>>>(pos, W_in, b_in, posqk);
  fused_attn<<<2048, 256, lds_bytes, stream>>>(x, posqk, b_in, b_out, Wih, Woh, out);
}
